// Attention_77309412187
// MI455X (gfx1250) — matrix operand order repaired and checked by reading, not compiled, not run on hardware
//
#include <hip/hip_runtime.h>
#include <hip/hip_bf16.h>

typedef _Float16 half_t;
typedef __attribute__((ext_vector_type(16))) _Float16 v16h;
typedef __attribute__((ext_vector_type(8)))  _Float16 v8h;
typedef __attribute__((ext_vector_type(8)))  float    v8f;

#define NB 16
#define NC 64
#define NPIX 4096    // 64*64
#define NPOOL 1024   // 32*32
#define CK 8         // C/8
#define CG 32        // C/2

// ---------------------------------------------------------------------------
// Kernel 1: theta = conv1x1(x, W_theta) + b   -> f16 [B][N][8]
// One thread per (b, n). Lanes adjacent in n -> coalesced per-channel reads.
// ---------------------------------------------------------------------------
__global__ __launch_bounds__(256) void proj_theta_kernel(
    const float* __restrict__ x,      // [B][64][N]
    const float* __restrict__ Wt,     // [8][64]
    const float* __restrict__ bt,     // [8]
    half_t* __restrict__ theta_h)     // [B][N][8]
{
  const int gid = blockIdx.x * blockDim.x + threadIdx.x;   // 0..65535
  const int b = gid >> 12;
  const int n = gid & (NPIX - 1);
  const float* xp = x + (size_t)b * NC * NPIX + n;

  float acc[CK];
#pragma unroll
  for (int o = 0; o < CK; ++o) acc[o] = bt[o];
  for (int c = 0; c < NC; ++c) {
    const float xv = xp[(size_t)c * NPIX];
#pragma unroll
    for (int o = 0; o < CK; ++o) acc[o] = fmaf(Wt[o * NC + c], xv, acc[o]);
  }
  v8h th;
#pragma unroll
  for (int o = 0; o < CK; ++o) th[o] = (half_t)acc[o];
  *reinterpret_cast<v8h*>(theta_h + (size_t)gid * CK) = th;
}

// ---------------------------------------------------------------------------
// Kernel 2: phi = maxpool2(conv(x,W_phi)) -> f16 [B][M][8]
//           g   = maxpool2(conv(x,W_g))   -> f16 [B][32][M]  (c-major!)
// One thread per pooled cell (b, m).
// ---------------------------------------------------------------------------
__global__ __launch_bounds__(256) void proj_phig_kernel(
    const float* __restrict__ x,      // [B][64][N]
    const float* __restrict__ Wp,     // [8][64]
    const float* __restrict__ bp,     // [8]
    const float* __restrict__ Wg,     // [32][64]
    const float* __restrict__ bg,     // [32]
    half_t* __restrict__ phi_h,       // [B][M][8]
    half_t* __restrict__ g_h)         // [B][32][M]
{
  const int gid = blockIdx.x * blockDim.x + threadIdx.x;   // 0..16383
  const int b = gid >> 10;
  const int m = gid & (NPOOL - 1);
  const int h2 = m >> 5, w2 = m & 31;

  float pmax[CK], gmax[CG];
#pragma unroll
  for (int o = 0; o < CK; ++o) pmax[o] = -3.0e38f;
#pragma unroll
  for (int o = 0; o < CG; ++o) gmax[o] = -3.0e38f;

  for (int dy = 0; dy < 2; ++dy) {
    for (int dx = 0; dx < 2; ++dx) {
      const int n = (2 * h2 + dy) * 64 + (2 * w2 + dx);
      const float* xp = x + (size_t)b * NC * NPIX + n;
      float ap[CK], ag[CG];
#pragma unroll
      for (int o = 0; o < CK; ++o) ap[o] = bp[o];
#pragma unroll
      for (int o = 0; o < CG; ++o) ag[o] = bg[o];
      for (int c = 0; c < NC; ++c) {
        const float xv = xp[(size_t)c * NPIX];
#pragma unroll
        for (int o = 0; o < CK; ++o) ap[o] = fmaf(Wp[o * NC + c], xv, ap[o]);
#pragma unroll
        for (int o = 0; o < CG; ++o) ag[o] = fmaf(Wg[o * NC + c], xv, ag[o]);
      }
#pragma unroll
      for (int o = 0; o < CK; ++o) pmax[o] = fmaxf(pmax[o], ap[o]);
#pragma unroll
      for (int o = 0; o < CG; ++o) gmax[o] = fmaxf(gmax[o], ag[o]);
    }
  }
  v8h ph;
#pragma unroll
  for (int o = 0; o < CK; ++o) ph[o] = (half_t)pmax[o];
  *reinterpret_cast<v8h*>(phi_h + (size_t)gid * CK) = ph;
#pragma unroll
  for (int o = 0; o < CG; ++o)
    g_h[(size_t)b * CG * NPOOL + (size_t)o * NPOOL + m] = (half_t)gmax[o];
}

// ---------------------------------------------------------------------------
// Kernel 3: flash attention, one wave per 16-query tile.
//   S^T  = phi_tile(A, K=8 zero-padded to 32) x theta(B)      -> n on lanes
//   P    = exp(S^T - rowmax)   (C-layout == next A-layout, no shuffles)
//   oacc = P(A) x g(B), 32 channels = 2 accum tiles, flash-rescaled
// Output: o_buf f16 [B][N][32]
// ---------------------------------------------------------------------------
__global__ __launch_bounds__(256) void attn_kernel(
    const half_t* __restrict__ theta_h,  // [B][N][8]
    const half_t* __restrict__ phi_h,    // [B][M][8]
    const half_t* __restrict__ g_h,      // [B][32][M]
    half_t* __restrict__ o_buf)          // [B][N][32]
{
  const int wave = blockIdx.x * (blockDim.x >> 5) + (threadIdx.x >> 5);
  const int lane = threadIdx.x & 31;
  const int b  = wave >> 8;            // 256 tiles per batch
  const int n0 = (wave & 255) << 4;    // query tile base
  const int lo = lane & 15;
  const bool hi = lane >= 16;

  // B-fragment of theta [K=c x N=n]: lanes<16 hold K=0..15 (c0..7 real, rest 0)
  v16h bt;
#pragma unroll
  for (int i = 0; i < 16; ++i) bt[i] = (half_t)0.0f;
  if (!hi) {
    const v8h t = *reinterpret_cast<const v8h*>(
        theta_h + ((size_t)b * NPIX + n0 + lo) * CK);
#pragma unroll
    for (int i = 0; i < CK; ++i) bt[i] = t[i];
  }

  v8f oacc0, oacc1;
#pragma unroll
  for (int i = 0; i < 8; ++i) { oacc0[i] = 0.0f; oacc1[i] = 0.0f; }
  float mrun = -3.0e38f, lrun = 0.0f;

  const half_t* phib = phi_h + (size_t)b * NPOOL * CK;
  const half_t* gb   = g_h + (size_t)b * CG * NPOOL;

  for (int ch = 0; ch < NPOOL / 32; ++ch) {
    const int mb = ch * 32;
    if (ch + 1 < NPOOL / 32) {  // warm L2/WGP$ for the next key chunk
      __builtin_prefetch(phib + (size_t)(mb + 32) * CK, 0, 3);
      __builtin_prefetch(gb + (size_t)lo * NPOOL + mb + 32, 0, 3);
    }

    // A-fragments of phi: rows m (lanes<16), K=c zero-padded
    v16h ap1, ap2;
#pragma unroll
    for (int i = 0; i < 16; ++i) { ap1[i] = (half_t)0.0f; ap2[i] = (half_t)0.0f; }
    if (!hi) {
      const v8h p1 = *reinterpret_cast<const v8h*>(phib + (size_t)(mb + lo) * CK);
      const v8h p2 = *reinterpret_cast<const v8h*>(phib + (size_t)(mb + 16 + lo) * CK);
#pragma unroll
      for (int i = 0; i < CK; ++i) { ap1[i] = p1[i]; ap2[i] = p2[i]; }
    }

    // S^T tiles [m x n]: lane column = query n, rows m over regs/lane-halves
    v8f s1, s2;
#pragma unroll
    for (int i = 0; i < 8; ++i) { s1[i] = 0.0f; s2[i] = 0.0f; }
    s1 = __builtin_amdgcn_wmma_f32_16x16x32_f16(false, ap1, false, bt, (short)0, s1, false, false);
    s2 = __builtin_amdgcn_wmma_f32_16x16x32_f16(false, ap2, false, bt, (short)0, s2, false, false);

    // online softmax stats for query n = lane&15 (partner = lane^16)
    float cmax = s1[0];
#pragma unroll
    for (int i = 0; i < 8; ++i) { cmax = fmaxf(cmax, s1[i]); cmax = fmaxf(cmax, s2[i]); }
    cmax = fmaxf(cmax, __shfl_xor(cmax, 16, 32));
    const float mnew = fmaxf(mrun, cmax);

    float p1r[8], p2r[8], rs = 0.0f;
#pragma unroll
    for (int i = 0; i < 8; ++i) {
      p1r[i] = __expf(s1[i] - mnew);
      p2r[i] = __expf(s2[i] - mnew);
      rs += p1r[i] + p2r[i];
    }
    rs += __shfl_xor(rs, 16, 32);
    const float corr = __expf(mrun - mnew);
    lrun = lrun * corr + rs;
    mrun = mnew;

    // rescale o accumulators: row n = reg + (hi?8:0) -> broadcast corr[n]
#pragma unroll
    for (int r = 0; r < 8; ++r) {
      const float cr = __shfl(corr, r + (hi ? 8 : 0), 32);
      oacc0[r] *= cr;
      oacc1[r] *= cr;
    }

    // P as A-fragment [n x m]: C-layout maps 1:1 onto A-layout per lane
    v16h apP;
#pragma unroll
    for (int i = 0; i < 8; ++i) { apP[i] = (half_t)p1r[i]; apP[i + 8] = (half_t)p2r[i]; }

    // g B-fragments [K=m x N=c]: two 8-f16 runs per lane, m {8h..8h+7, 16+8h..} (c-major storage)
    const half_t* gl = gb + (size_t)lo * NPOOL + mb + (hi ? 8 : 0);
    v16h bg0, bg1;
    {
      const v8h a0 = *reinterpret_cast<const v8h*>(gl);
      const v8h a1 = *reinterpret_cast<const v8h*>(gl + 16);
      const v8h c0 = *reinterpret_cast<const v8h*>(gl + 16 * NPOOL);
      const v8h c1 = *reinterpret_cast<const v8h*>(gl + 16 * NPOOL + 16);
#pragma unroll
      for (int i = 0; i < 8; ++i) {
        bg0[i] = a0[i]; bg0[i + 8] = a1[i];
        bg1[i] = c0[i]; bg1[i + 8] = c1[i];
      }
    }
    oacc0 = __builtin_amdgcn_wmma_f32_16x16x32_f16(false, apP, false, bg0, (short)0, oacc0, false, false);
    oacc1 = __builtin_amdgcn_wmma_f32_16x16x32_f16(false, apP, false, bg1, (short)0, oacc1, false, false);
  }

  // finalize: divide by row sum, store [B][N][32] f16
  const float linv = 1.0f / lrun;
  const size_t obase = ((size_t)b * NPIX + n0) * CG;
#pragma unroll
  for (int r = 0; r < 8; ++r) {
    const float li = __shfl(linv, r + (hi ? 8 : 0), 32);
    const int n = r + (hi ? 8 : 0);
    o_buf[obase + (size_t)n * CG + lo]      = (half_t)(oacc0[r] * li);
    o_buf[obase + (size_t)n * CG + 16 + lo] = (half_t)(oacc1[r] * li);
  }
}

// ---------------------------------------------------------------------------
// Kernel 4: out = gamma * (o @ W_o^T + b_o) + x ; one wave per 16-query tile,
// 4 WMMAs cover the 64 output channels.
// ---------------------------------------------------------------------------
__global__ __launch_bounds__(256) void outconv_kernel(
    const half_t* __restrict__ o_buf,  // [B][N][32]
    const float* __restrict__ Wo,      // [64][32]
    const float* __restrict__ bo,      // [64]
    const float* __restrict__ x,       // [B][64][N]
    const float* __restrict__ gamma,   // [1]
    float* __restrict__ out)           // [B][64][N]
{
  const int wave = blockIdx.x * (blockDim.x >> 5) + (threadIdx.x >> 5);
  const int lane = threadIdx.x & 31;
  const int b  = wave >> 8;
  const int n0 = (wave & 255) << 4;
  const int lo = lane & 15;
  const bool hi = lane >= 16;
  const float gm = gamma[0];

  // A-fragment of o [n x ci]: lane<16 needs ci {0..7,16..23}, lane>=16 {8..15,24..31}
  const half_t* ob = o_buf + ((size_t)b * NPIX + n0 + lo) * CG + (hi ? 8 : 0);
  v16h a;
  {
    const v8h a0 = *reinterpret_cast<const v8h*>(ob);
    const v8h a1 = *reinterpret_cast<const v8h*>(ob + 16);
#pragma unroll
    for (int i = 0; i < 8; ++i) { a[i] = a0[i]; a[i + 8] = a1[i]; }
  }

#pragma unroll
  for (int ct = 0; ct < 4; ++ct) {
    const int co = ct * 16 + lo;
    const float* wp = Wo + co * CG + (hi ? 8 : 0);
    v16h bw;
#pragma unroll
    for (int i = 0; i < 16; ++i) bw[i] = (half_t)wp[i < 8 ? i : i + 8];

    v8f d;
#pragma unroll
    for (int i = 0; i < 8; ++i) d[i] = 0.0f;
    d = __builtin_amdgcn_wmma_f32_16x16x32_f16(false, a, false, bw, (short)0, d, false, false);

    const float bias = bo[co];
    const float* xr = x   + ((size_t)b * NC + co) * NPIX + n0;
    float* outr     = out + ((size_t)b * NC + co) * NPIX + n0;
#pragma unroll
    for (int r = 0; r < 8; ++r) {
      const int n = r + (hi ? 8 : 0);
      outr[n] = fmaf(gm, d[r] + bias, xr[n]);
    }
  }
}

// ---------------------------------------------------------------------------
extern "C" void kernel_launch(void* const* d_in, const int* in_sizes, int n_in,
                              void* d_out, int out_size, void* d_ws, size_t ws_size,
                              hipStream_t stream) {
  const float* x      = (const float*)d_in[0];
  const float* Wt     = (const float*)d_in[1];
  const float* bt     = (const float*)d_in[2];
  const float* Wp     = (const float*)d_in[3];
  const float* bp     = (const float*)d_in[4];
  const float* Wg     = (const float*)d_in[5];
  const float* bg     = (const float*)d_in[6];
  const float* Wo     = (const float*)d_in[7];
  const float* bo     = (const float*)d_in[8];
  const float* gamma  = (const float*)d_in[9];
  float* out = (float*)d_out;

  char* ws = (char*)d_ws;
  half_t* theta_h = (half_t*)(ws);                       // 16*4096*8*2  = 1 MiB
  half_t* phi_h   = (half_t*)(ws + (1u << 20));          // 16*1024*8*2  = 256 KiB
  half_t* g_h     = (half_t*)(ws + (1u << 20) + (256u << 10));  // 16*32*1024*2 = 1 MiB
  half_t* o_buf   = (half_t*)(ws + (1u << 20) + (256u << 10) + (1u << 20)); // 4 MiB

  proj_theta_kernel<<<256, 256, 0, stream>>>(x, Wt, bt, theta_h);
  proj_phig_kernel<<<64, 256, 0, stream>>>(x, Wp, bp, Wg, bg, phi_h, g_h);
  attn_kernel<<<512, 256, 0, stream>>>(theta_h, phi_h, g_h, o_buf);
  outconv_kernel<<<512, 256, 0, stream>>>(o_buf, Wo, bo, x, gamma, out);
}
